// SubtractiveVocoder_87144886436587
// MI455X (gfx1250) — compile-verified, matrix operand order checked by reading, not run on hardware
//
#include <hip/hip_runtime.h>
#include <hip/hip_bf16.h>
#include <math.h>

// ---------------- problem constants ----------------
#define N_MELS  80
#define SRATE   24000.0f
#define HOP     256
#define NFFT    1024
#define FBIN    513           // NFFT/2+1
#define BATCH   16
#define FRAMES  800
#define TLEN    (FRAMES*HOP)  // 204800
#define NFR     801           // STFT frames
#define LLEN    (HOP*(NFR-1)+NFFT) // 205824
#define RLEN    8192          // reverb taps
#define VCLEN   256           // vocal-cord taps

// ---------------- workspace layout (floats) ----------------
#define OFF_F0UP      ((size_t)0)
#define OFF_SAW       (OFF_F0UP + (size_t)BATCH*TLEN)            // reused later for voi_time
#define OFF_IMP       (OFF_SAW + (size_t)BATCH*TLEN)
#define OFF_IMPC      (OFF_IMP + (size_t)BATCH*TLEN)
#define OFF_STFT_IMP  (OFF_IMPC + (size_t)BATCH*TLEN)            // B*FBIN*NFR complex
#define OFF_STFT_NOI  (OFF_STFT_IMP + (size_t)BATCH*FBIN*NFR*2)
#define OFF_KIMP_AMP  (OFF_STFT_NOI + (size_t)BATCH*FBIN*NFR*2)  // B*FBIN*FRAMES real
#define OFF_KNOI_AMP  (OFF_KIMP_AMP + (size_t)BATCH*FBIN*FRAMES)
#define OFF_KIMP_C    (OFF_KNOI_AMP + (size_t)BATCH*FBIN*FRAMES) // complex
#define OFF_Y         (OFF_KIMP_C + (size_t)BATCH*FBIN*FRAMES*2)

typedef __attribute__((ext_vector_type(16))) __bf16 v16bf;
typedef __attribute__((ext_vector_type(8)))  float  v8f;

// ============ Tensor Data Mover (gfx1250) ============
#if __has_builtin(__builtin_amdgcn_tensor_load_to_lds)
#define HAVE_TDM 1
typedef unsigned int uint32x4 __attribute__((ext_vector_type(4)));
typedef int          int32x8  __attribute__((ext_vector_type(8)));
typedef int          int32x4  __attribute__((ext_vector_type(4)));

// 2-D f32 tile load, LDS rows padded by 1 DWORD every 64 DWORDs (stride 65)
__device__ __forceinline__ void tdm_load_2d_pad64(unsigned lds_addr, const float* gptr,
                                                  int tensor_d0, int tensor_d1,
                                                  int tile_d0, int tile_d1,
                                                  int row_stride) {
  unsigned long long ga = (unsigned long long)(uintptr_t)gptr;
  uint32x4 g0;
  g0[0] = 1u;                                   // count=1, user mode
  g0[1] = lds_addr;                             // LDS byte address
  g0[2] = (unsigned)(ga & 0xffffffffull);       // global_addr[31:0]
  g0[3] = (unsigned)((ga >> 32) & 0x1ffffffull) // global_addr[56:32]
        | (2u << 30);                           // type=2 (image)
  int32x8 g1;
  g1[0] = (int)((2u << 16)                      // data_size = 4 bytes
        | (1u << 20)                            // pad_enable
        | (5u << 22));                          // pad_interval: 64 DWORDs; pad_amount=0 -> 1 DWORD
  g1[1] = (int)(((unsigned)tensor_d0 & 0xffffu) << 16);               // tensor_dim0[15:0]
  g1[2] = (int)((((unsigned)tensor_d0 >> 16) & 0xffffu)
        | (((unsigned)tensor_d1 & 0xffffu) << 16));                   // dim0[31:16] | dim1[15:0]
  g1[3] = (int)((((unsigned)tensor_d1 >> 16) & 0xffffu)
        | (((unsigned)tile_d0 & 0xffffu) << 16));                     // dim1[31:16] | tile_dim0
  g1[4] = (int)((unsigned)tile_d1 & 0xffffu);                         // tile_dim1 (tile_dim2=0)
  g1[5] = (int)row_stride;                                            // tensor_dim0_stride[31:0]
  g1[6] = 0;
  g1[7] = 0;
  int32x4 gz4 = {0, 0, 0, 0};
  int32x8 gz8 = {0, 0, 0, 0, 0, 0, 0, 0};
  // 6-arg (clang-23/therock) form: groups 0..3 + extra group + cache policy
  __builtin_amdgcn_tensor_load_to_lds(g0, g1, gz4, gz4, gz8, 0);
}
#else
#define HAVE_TDM 0
#endif

// =====================================================================
// 1. f0 upsample + running-phase scan
// =====================================================================
__global__ void __launch_bounds__(1024) scan_kernel(const float* __restrict__ f0,
                                                    float* __restrict__ f0_up,
                                                    float* __restrict__ saw) {
  __shared__ float sh[1024];
  __shared__ float carry;
  const int b = blockIdx.x;
  const int tid = threadIdx.x;
  if (tid == 0) carry = 0.f;
  __syncthreads();
  for (int base = 0; base < TLEN; base += 1024) {
    const int n = base + tid;
    float pos = (n + 0.5f) * (1.0f / HOP) - 0.5f;
    pos = fminf(fmaxf(pos, 0.f), (float)(FRAMES - 1));
    int i0 = (int)floorf(pos);
    int i1 = min(i0 + 1, FRAMES - 1);
    float w = pos - (float)i0;
    float v = f0[b*FRAMES + i0] * (1.f - w) + f0[b*FRAMES + i1] * w;
    f0_up[(size_t)b*TLEN + n] = v;
    sh[tid] = v;
    __syncthreads();
    for (int off = 1; off < 1024; off <<= 1) {
      float t = (tid >= off) ? sh[tid - off] : 0.f;
      __syncthreads();
      sh[tid] += t;
      __syncthreads();
    }
    float cum = sh[tid] + carry;
    saw[(size_t)b*TLEN + n] = fmodf(cum * (1.0f / SRATE), 1.0f);
    __syncthreads();
    if (tid == 1023) carry += sh[1023];
    __syncthreads();
  }
}

// =====================================================================
// 2. impulse train
// =====================================================================
__global__ void impulse_kernel(const float* __restrict__ f0_up,
                               const float* __restrict__ saw,
                               float* __restrict__ imp) {
  int i = blockIdx.x * blockDim.x + threadIdx.x;
  if (i >= BATCH * TLEN) return;
  int b = i / TLEN, n = i % TLEN;
  float s0 = saw[i];
  float s1 = saw[(size_t)b*TLEN + ((n + 1) % TLEN)];
  float pulse = (s0 - s1 > 0.5f) ? 1.f : 0.f;
  float sc = rsqrtf(fmaxf(f0_up[i], 20.f)) * 154.9193338482967f;
  imp[i] = pulse * sc;
}

// =====================================================================
// 3. 256-tap causal FIR (vocal cord)
// =====================================================================
__global__ void __launch_bounds__(256) vocal_fir_kernel(const float* __restrict__ sig,
                                                        const float* __restrict__ vc,
                                                        float* __restrict__ out) {
  __shared__ float kb[VCLEN];
  __shared__ float sb[512];
  const int b = blockIdx.y, n0 = blockIdx.x * 256, tid = threadIdx.x;
  kb[tid] = vc[b*VCLEN + tid];
  const int base = n0 - 255;
  int g = base + tid;
  sb[tid] = (g >= 0) ? sig[(size_t)b*TLEN + g] : 0.f;
  if (tid < 255) {
    g = base + 256 + tid;
    sb[256 + tid] = (g < TLEN) ? sig[(size_t)b*TLEN + g] : 0.f;
  }
  __syncthreads();
  float acc = 0.f;
  #pragma unroll 8
  for (int j = 0; j < VCLEN; ++j) acc += sb[255 + tid - j] * kb[j];
  out[(size_t)b*TLEN + n0 + tid] = acc;
}

// =====================================================================
// shared-memory radix-2 1024-pt FFT (256 threads)
// =====================================================================
__device__ __forceinline__ void fft1024_shared(float* re, float* im, int tid, float sgn) {
  for (int s = 1; s <= 10; ++s) {
    const int half = 1 << (s - 1);
    const int len  = half << 1;
    for (int k = tid; k < 512; k += 256) {
      int grp = k >> (s - 1);
      int pos = k & (half - 1);
      int i0 = grp * len + pos;
      int i1 = i0 + half;
      float ang = sgn * 6.283185307179586f * (float)pos / (float)len;
      float wi, wr;
      __sincosf(ang, &wi, &wr);
      float xr = re[i1], xi = im[i1];
      float tr = xr * wr - xi * wi;
      float ti = xr * wi + xi * wr;
      float ur = re[i0], ui = im[i0];
      re[i0] = ur + tr; im[i0] = ui + ti;
      re[i1] = ur - tr; im[i1] = ui - ti;
    }
    __syncthreads();
  }
}

// =====================================================================
// 4. STFT
// =====================================================================
__global__ void __launch_bounds__(256) stft_kernel(const float* __restrict__ imp_sig,
                                                   const float* __restrict__ noise,
                                                   const float* __restrict__ win,
                                                   float* __restrict__ stft_imp,
                                                   float* __restrict__ stft_noi) {
  __shared__ float re[NFFT], im[NFFT], xr[NFFT];
  const int f = blockIdx.x, b = blockIdx.y, z = blockIdx.z, tid = threadIdx.x;
  const float* src = z ? noise : imp_sig;
  const float scale = z ? 0.33333f : 1.f;
  for (int n = tid; n < NFFT; n += 256) {
    int p = f * HOP + n - NFFT/2;
    if (p < 0) p = -p;
    if (p >= TLEN) p = 2*TLEN - 2 - p;
    xr[n] = src[(size_t)b*TLEN + p] * scale * win[n];
  }
  __syncthreads();
  for (int i = tid; i < NFFT; i += 256) {
    int j = __brev((unsigned)i) >> 22;
    re[i] = xr[j]; im[i] = 0.f;
  }
  __syncthreads();
  fft1024_shared(re, im, tid, -1.f);
  float* out = z ? stft_noi : stft_imp;
  for (int k = tid; k < FBIN; k += 256) {
    size_t idx = (((size_t)b*FBIN + k)*NFR + f) * 2;
    out[idx] = re[k]; out[idx + 1] = im[k];
  }
}

// =====================================================================
// 5. einsum 'fm,bmt->bft' + ReLU as bf16 WMMA GEMM
//    block = 4 wave32s; output tile 16(M) x 64(N); K=80 padded to 96.
//    B tiles staged to LDS by the Tensor Data Mover (pad_interval=64 DW,
//    pad_amount=1 DW -> LDS row stride 65 for conflict-free column reads).
// =====================================================================
#define BROW 65           // padded LDS row stride (floats) for B tiles
#define AROW 97           // padded LDS row stride (floats) for A tile
__global__ void __launch_bounds__(128) envelope_gemm_wmma(const float* __restrict__ inv_mel,
                                                          const float* __restrict__ env_imp,
                                                          const float* __restrict__ env_noi,
                                                          float* __restrict__ kimp,
                                                          float* __restrict__ knoi) {
  __shared__ float aT[16 * AROW];       //  6.2 KB
  __shared__ float bI[96 * BROW];       // 25.0 KB
  __shared__ float bN[96 * BROW];       // 25.0 KB
  const int n0    = blockIdx.x * 64;    // column base (t)
  const int mTile = blockIdx.y;
  const int b     = blockIdx.z;
  const int tid   = threadIdx.x;
  const int wave  = tid >> 5, lane = tid & 31;
  const int c  = lane & 15;
  const int hi = lane >> 4;
  const int mBase = mTile * 16;

  // ---- stage A tile (16 x 96), branch-free: clamped load * mask ----
  for (int idx = tid; idx < 16 * AROW; idx += 128) {
    int m = idx / AROW, k = idx - m * AROW;
    int gm = mBase + m;
    float v = inv_mel[min(gm, FBIN - 1) * N_MELS + min(k, N_MELS - 1)];
    aT[idx] = (gm < FBIN && k < N_MELS) ? v : 0.f;
  }
  // ---- zero-fill K rows 80..95 of the B tiles ----
  for (int idx = tid; idx < 16 * BROW; idx += 128) {
    bI[80 * BROW + idx] = 0.f;
    bN[80 * BROW + idx] = 0.f;
  }
#if HAVE_TDM
  if (wave == 0) {
    const float* gI = env_imp + (size_t)b * N_MELS * FRAMES + n0;
    const float* gN = env_noi + (size_t)b * N_MELS * FRAMES + n0;
    tdm_load_2d_pad64((unsigned)(uintptr_t)&bI[0], gI, FRAMES, N_MELS, 64, N_MELS, FRAMES);
    tdm_load_2d_pad64((unsigned)(uintptr_t)&bN[0], gN, FRAMES, N_MELS, 64, N_MELS, FRAMES);
    __builtin_amdgcn_s_wait_tensorcnt(0);
  }
#else
  for (int idx = tid; idx < N_MELS * 64; idx += 128) {
    int k = idx >> 6, n = idx & 63;
    int t = n0 + n;
    float mask = (t < FRAMES) ? 1.f : 0.f;
    size_t g = ((size_t)b * N_MELS + k) * FRAMES + min(t, FRAMES - 1);
    bI[k * BROW + n] = env_imp[g] * mask;
    bN[k * BROW + n] = env_noi[g] * mask;
  }
#endif
  __syncthreads();

  // ---- fragments from LDS (unconditional reads) + WMMA ----
  const int nLocal = wave * 16 + c;
  v8f accI = {}; v8f accN = {};
  #pragma unroll
  for (int kk = 0; kk < 3; ++kk) {
    v16bf a, fbi, fbn;
    #pragma unroll
    for (int j = 0; j < 16; ++j) {
      // 16-bit A/B striping: lanes 0-15 hold K {0..7,16..23}, lanes 16-31 K {8..15,24..31}
      int k = kk * 32 + ((j < 8) ? j : j + 8) + hi * 8;
      a[j]   = (__bf16)aT[c * AROW + k];
      fbi[j] = (__bf16)bI[k * BROW + nLocal];
      fbn[j] = (__bf16)bN[k * BROW + nLocal];
    }
    accI = __builtin_amdgcn_wmma_f32_16x16x32_bf16(false, a, false, fbi, (short)0, accI, false, false);
    accN = __builtin_amdgcn_wmma_f32_16x16x32_bf16(false, a, false, fbn, (short)0, accN, false, false);
  }
  // D layout: VGPR r, lanes 0-15 -> M=r, lanes 16-31 -> M=8+r; N = lane&15
  const int tcol = n0 + nLocal;
  #pragma unroll
  for (int r = 0; r < 8; ++r) {
    int mr = mBase + r + hi * 8;
    if (mr < FBIN && tcol < FRAMES) {
      size_t idx = ((size_t)b * FBIN + mr) * FRAMES + tcol;
      kimp[idx] = fmaxf(accI[r], 0.f);
      knoi[idx] = fmaxf(accN[r], 0.f);
    }
  }
}

// =====================================================================
// 6. minimum phase
// =====================================================================
__global__ void __launch_bounds__(256) minphase_kernel(const float* __restrict__ amp,
                                                       float* __restrict__ outc) {
  __shared__ float re[NFFT], im[NFFT], tmp[NFFT];
  __shared__ float la[FBIN];
  const int t = blockIdx.x, b = blockIdx.y, tid = threadIdx.x;
  for (int k = tid; k < FBIN; k += 256)
    la[k] = __logf(fmaxf(amp[((size_t)b*FBIN + k)*FRAMES + t], 1e-6f));
  __syncthreads();
  for (int i = tid; i < NFFT; i += 256) {
    int j = __brev((unsigned)i) >> 22;
    int jj = (j <= 512) ? j : NFFT - j;
    re[i] = la[jj]; im[i] = 0.f;
  }
  __syncthreads();
  fft1024_shared(re, im, tid, 1.f);
  for (int i = tid; i < NFFT; i += 256) {
    float foldv = (i == 0 || i == 512) ? 1.f : (i < 512 ? 2.f : 0.f);
    tmp[i] = re[i] * (1.f / NFFT) * foldv;
  }
  __syncthreads();
  for (int i = tid; i < NFFT; i += 256) {
    int j = __brev((unsigned)i) >> 22;
    re[i] = tmp[j]; im[i] = 0.f;
  }
  __syncthreads();
  fft1024_shared(re, im, tid, -1.f);
  for (int k = tid; k < FBIN; k += 256) {
    float er = __expf(re[k]);
    float sn, cs;
    __sincosf(im[k], &sn, &cs);
    size_t idx = (((size_t)b*FBIN + k)*FRAMES + t) * 2;
    outc[idx] = er * cs; outc[idx + 1] = er * sn;
  }
}

// =====================================================================
// 7. spectral mix
// =====================================================================
__global__ void mix_kernel(const float* __restrict__ f0,
                           const float* __restrict__ kimp_c,
                           const float* __restrict__ knoi_a,
                           float* __restrict__ stft_imp,
                           const float* __restrict__ stft_noi) {
  size_t i = (size_t)blockIdx.x * blockDim.x + threadIdx.x;
  const size_t total = (size_t)BATCH * FBIN * NFR;
  if (i >= total) return;
  int f = (int)(i % NFR);
  size_t bk = i / NFR;
  int b = (int)(bk / FBIN);
  float ir = stft_imp[2*i], ii = stft_imp[2*i + 1];
  float nr = stft_noi[2*i], ni = stft_noi[2*i + 1];
  float f0p = (f == 0) ? 0.f : f0[b*FRAMES + f - 1];
  float mask = (f0p < 20.f) ? 1.f : 0.f;
  ir += nr * mask; ii += ni * mask;
  float kr = 0.f, ki = 0.f, kn = 0.f;
  if (f > 0) {
    size_t j = bk * FRAMES + (f - 1);
    kr = kimp_c[2*j]; ki = kimp_c[2*j + 1];
    kn = knoi_a[j];
  }
  stft_imp[2*i]     = ir * kr - ii * ki + nr * kn;
  stft_imp[2*i + 1] = ir * ki + ii * kr + ni * kn;
}

// =====================================================================
// 8. iSTFT
// =====================================================================
__global__ void fill_zero_kernel(float* __restrict__ p, size_t n) {
  size_t i = (size_t)blockIdx.x * blockDim.x + threadIdx.x;
  if (i < n) p[i] = 0.f;
}

__global__ void __launch_bounds__(256) istft_kernel(const float* __restrict__ spec,
                                                    const float* __restrict__ win,
                                                    float* __restrict__ Y) {
  __shared__ float re[NFFT], im[NFFT];
  __shared__ float sr[FBIN], si[FBIN];
  const int f = blockIdx.x, b = blockIdx.y, tid = threadIdx.x;
  for (int k = tid; k < FBIN; k += 256) {
    size_t idx = (((size_t)b*FBIN + k)*NFR + f) * 2;
    sr[k] = spec[idx]; si[k] = spec[idx + 1];
  }
  __syncthreads();
  for (int i = tid; i < NFFT; i += 256) {
    int j = __brev((unsigned)i) >> 22;
    if (j <= 512) { re[i] = sr[j];        im[i] = si[j]; }
    else          { re[i] = sr[NFFT - j]; im[i] = -si[NFFT - j]; }
  }
  __syncthreads();
  fft1024_shared(re, im, tid, 1.f);
  for (int n = tid; n < NFFT; n += 256) {
    float v = re[n] * (1.f / NFFT) * win[n];
    atomicAdd(&Y[(size_t)b*LLEN + f*HOP + n], v);
  }
}

// =====================================================================
// 9. window-square normalization + crop
// =====================================================================
__global__ void normalize_kernel(const float* __restrict__ Y,
                                 const float* __restrict__ win,
                                 float* __restrict__ voi_t) {
  int i = blockIdx.x * blockDim.x + threadIdx.x;
  if (i >= BATCH * TLEN) return;
  int b = i / TLEN, n = i % TLEN;
  int gi = n + NFFT/2;
  float wsq = 0.f;
  int fmaxi = gi / HOP;
  #pragma unroll
  for (int d = 0; d < 4; ++d) {
    int f = fmaxi - d;
    int off = gi - f * HOP;
    if (f >= 0 && f < NFR && off < NFFT) { float w = win[off]; wsq += w * w; }
  }
  float y = Y[(size_t)b*LLEN + gi];
  voi_t[i] = y / (wsq > 1e-11f ? wsq : 1.f);
}

// =====================================================================
// 10. 8192-tap reverb FIR
// =====================================================================
__global__ void __launch_bounds__(256) reverb_fir_kernel(const float* __restrict__ v,
                                                         const float* __restrict__ rev,
                                                         float* __restrict__ out) {
  __shared__ float kb[256];
  __shared__ float sb[512];
  const int b = blockIdx.y, n0 = blockIdx.x * 256, tid = threadIdx.x;
  float acc = 0.f;
  for (int c = 0; c < RLEN; c += 256) {
    __syncthreads();
    kb[tid] = rev[b*RLEN + c + tid];
    const int base = n0 - c - 255;
    int g = base + tid;
    sb[tid] = (g >= 0 && g < TLEN) ? v[(size_t)b*TLEN + g] : 0.f;
    if (tid < 255) {
      g = base + 256 + tid;
      sb[256 + tid] = (g >= 0 && g < TLEN) ? v[(size_t)b*TLEN + g] : 0.f;
    }
    if (c + 256 < RLEN) __builtin_prefetch(&rev[b*RLEN + c + 256 + tid], 0, 3);
    __syncthreads();
    #pragma unroll 8
    for (int j = 0; j < 256; ++j) acc += sb[255 + tid - j] * kb[j];
  }
  out[(size_t)b*TLEN + n0 + tid] = acc;
}

// =====================================================================
extern "C" void kernel_launch(void* const* d_in, const int* in_sizes, int n_in,
                              void* d_out, int out_size, void* d_ws, size_t ws_size,
                              hipStream_t stream) {
  const float* f0       = (const float*)d_in[0];
  const float* env_imp  = (const float*)d_in[1];
  const float* env_noi  = (const float*)d_in[2];
  const float* vocal    = (const float*)d_in[3];
  const float* reverb   = (const float*)d_in[4];
  const float* noise    = (const float*)d_in[5];
  const float* hann     = (const float*)d_in[6];
  const float* inv_mel  = (const float*)d_in[7];

  float* ws    = (float*)d_ws;
  float* f0up  = ws + OFF_F0UP;
  float* saw   = ws + OFF_SAW;      // reused as voi_time later
  float* imp   = ws + OFF_IMP;
  float* impc  = ws + OFF_IMPC;
  float* stftI = ws + OFF_STFT_IMP;
  float* stftN = ws + OFF_STFT_NOI;
  float* kimpA = ws + OFF_KIMP_AMP;
  float* knoiA = ws + OFF_KNOI_AMP;
  float* kimpC = ws + OFF_KIMP_C;
  float* Y     = ws + OFF_Y;

  scan_kernel<<<BATCH, 1024, 0, stream>>>(f0, f0up, saw);
  impulse_kernel<<<(BATCH*TLEN + 255)/256, 256, 0, stream>>>(f0up, saw, imp);
  vocal_fir_kernel<<<dim3(TLEN/256, BATCH), 256, 0, stream>>>(imp, vocal, impc);
  stft_kernel<<<dim3(NFR, BATCH, 2), 256, 0, stream>>>(impc, noise, hann, stftI, stftN);

  envelope_gemm_wmma<<<dim3((FRAMES + 63)/64, (FBIN + 15)/16, BATCH), 128, 0, stream>>>(
      inv_mel, env_imp, env_noi, kimpA, knoiA);

  minphase_kernel<<<dim3(FRAMES, BATCH), 256, 0, stream>>>(kimpA, kimpC);

  {
    size_t total = (size_t)BATCH * FBIN * NFR;
    mix_kernel<<<(unsigned)((total + 255)/256), 256, 0, stream>>>(f0, kimpC, knoiA, stftI, stftN);
  }
  {
    size_t yn = (size_t)BATCH * LLEN;
    fill_zero_kernel<<<(unsigned)((yn + 255)/256), 256, 0, stream>>>(Y, yn);
  }
  istft_kernel<<<dim3(NFR, BATCH), 256, 0, stream>>>(stftI, hann, Y);
  normalize_kernel<<<(BATCH*TLEN + 255)/256, 256, 0, stream>>>(Y, hann, saw);
  reverb_fir_kernel<<<dim3(TLEN/256, BATCH), 256, 0, stream>>>(saw, reverb, (float*)d_out);
}
